// RNN_20005957665325
// MI455X (gfx1250) — compile-verified, matrix operand order checked by reading, not run on hardware
//
#include <hip/hip_runtime.h>

// ---------------- problem constants ----------------
#define BB     4
#define SS     2048
#define DIN    1024
#define DH     2048
#define DOUT   1024
#define PEDIM  16
#define DCAT   (DIN + PEDIM)      // 1040
#define KPAD   1056               // 33 * 32, zero-padded K for xc GEMMs
#define MROWS  (BB * SS)          // 8192

typedef __attribute__((ext_vector_type(16))) __bf16 v16bf;
typedef __attribute__((ext_vector_type(8)))  __bf16 v8bf;
typedef __attribute__((ext_vector_type(8)))  float  v8f;

// ---------------- pack xc = [x | PE | 0pad] into bf16, row stride KPAD ----------------
__global__ __launch_bounds__(256) void pack_x_kernel(const float* __restrict__ x,
                                                     __bf16* __restrict__ xcb) {
    int id = blockIdx.x * 256 + threadIdx.x;            // over MROWS * KPAD
    if (id >= MROWS * KPAD) return;
    int m = id / KPAD;
    int c = id - m * KPAD;
    int s = m & (SS - 1);                               // m = b*SS + s
    float v;
    if (c < DIN) {
        v = x[(size_t)m * DIN + c];
    } else if (c < DCAT) {
        int j = c - DIN;                                // 0..15
        int pair = j >> 1;
        float freq = __expf(-(float)(2 * pair) * (logf(10000.0f) / (float)PEDIM));
        float ang = (float)s * freq;
        v = ((j & 1) ? __cosf(ang) : __sinf(ang)) * 0.1f;
    } else {
        v = 0.0f;                                       // K pad
    }
    xcb[id] = (__bf16)v;
}

// ---------------- W[K,N] (fp32) -> Wt[n, kpad] (bf16, zero K-pad) ----------------
__global__ __launch_bounds__(256) void pack_w_kernel(const float* __restrict__ W,
                                                     __bf16* __restrict__ Wt,
                                                     int K, int Kpad, int N) {
    int id = blockIdx.x * 256 + threadIdx.x;            // over N * Kpad
    if (id >= N * Kpad) return;
    int n = id / Kpad;
    int k = id - n * Kpad;
    float v = (k < K) ? W[(size_t)k * N + n] : 0.0f;
    Wt[id] = (__bf16)v;
}

// ---------------- bf16 WMMA GEMM: out[M,N] = A[M,Kpad] @ Wt[N,Kpad]^T (+epilogue) --
// mode 0: out = acc + bias                      (dx -> d_out, fp32)
// mode 1: out = (acc + bias) * sqrt(1-a^2)      (xh -> ws, fp32; extra = a_logit[N])
// mode 2: out = (acc + bias + extra[m,n]) * 0.5 (final out; extra = dx in d_out)
__global__ __launch_bounds__(256) void gemm_bf16_kernel(const __bf16* __restrict__ A,
                                                        const __bf16* __restrict__ Bt,
                                                        int Kpad, int N,
                                                        float* __restrict__ out,
                                                        const float* __restrict__ bias,
                                                        const float* __restrict__ extra,
                                                        int mode) {
    const int lane = threadIdx.x & 31;
    const int wave = threadIdx.x >> 5;
    const int row  = lane & 15;        // fragment row (A) / fragment col-row (B^T)
    const int half = lane >> 4;        // lane half selects K sub-block
    const int n0 = blockIdx.x * 64;
    const int m0 = blockIdx.y * 128 + wave * 16;

    const __bf16* ap = A + (size_t)(m0 + row) * Kpad + 8 * half;
    const __bf16* bp0 = Bt + (size_t)(n0 +  0 + row) * Kpad + 8 * half;
    const __bf16* bp1 = Bt + (size_t)(n0 + 16 + row) * Kpad + 8 * half;
    const __bf16* bp2 = Bt + (size_t)(n0 + 32 + row) * Kpad + 8 * half;
    const __bf16* bp3 = Bt + (size_t)(n0 + 48 + row) * Kpad + 8 * half;

    v8f acc0 = {}, acc1 = {}, acc2 = {}, acc3 = {};

    for (int k0 = 0; k0 < Kpad; k0 += 32) {
        __builtin_prefetch(ap + k0 + 128, 0, 1);        // global_prefetch_b8 on A stream
        v8bf alo = *(const v8bf*)(ap + k0);
        v8bf ahi = *(const v8bf*)(ap + k0 + 16);
        v16bf af = __builtin_shufflevector(alo, ahi, 0,1,2,3,4,5,6,7,8,9,10,11,12,13,14,15);

        v8bf blo, bhi; v16bf bf;
        blo = *(const v8bf*)(bp0 + k0); bhi = *(const v8bf*)(bp0 + k0 + 16);
        bf = __builtin_shufflevector(blo, bhi, 0,1,2,3,4,5,6,7,8,9,10,11,12,13,14,15);
        acc0 = __builtin_amdgcn_wmma_f32_16x16x32_bf16(false, af, false, bf, (short)0, acc0, false, false);

        blo = *(const v8bf*)(bp1 + k0); bhi = *(const v8bf*)(bp1 + k0 + 16);
        bf = __builtin_shufflevector(blo, bhi, 0,1,2,3,4,5,6,7,8,9,10,11,12,13,14,15);
        acc1 = __builtin_amdgcn_wmma_f32_16x16x32_bf16(false, af, false, bf, (short)0, acc1, false, false);

        blo = *(const v8bf*)(bp2 + k0); bhi = *(const v8bf*)(bp2 + k0 + 16);
        bf = __builtin_shufflevector(blo, bhi, 0,1,2,3,4,5,6,7,8,9,10,11,12,13,14,15);
        acc2 = __builtin_amdgcn_wmma_f32_16x16x32_bf16(false, af, false, bf, (short)0, acc2, false, false);

        blo = *(const v8bf*)(bp3 + k0); bhi = *(const v8bf*)(bp3 + k0 + 16);
        bf = __builtin_shufflevector(blo, bhi, 0,1,2,3,4,5,6,7,8,9,10,11,12,13,14,15);
        acc3 = __builtin_amdgcn_wmma_f32_16x16x32_bf16(false, af, false, bf, (short)0, acc3, false, false);
    }

    // Epilogue. C/D layout: VGPR v, lane L -> m = v + 8*(L>>4), n = L&15.
    const int ncol = lane & 15;
    v8f accs[4] = {acc0, acc1, acc2, acc3};
    #pragma unroll
    for (int nt = 0; nt < 4; ++nt) {
        int n = n0 + nt * 16 + ncol;
        float bb = bias[n];
        float g = 1.0f;
        if (mode == 1) {
            float a = 1.0f / (1.0f + __expf(-extra[n]));     // sigmoid(a_logit)
            g = sqrtf(fmaxf(0.0f, 1.0f - a * a));
        }
        #pragma unroll
        for (int v = 0; v < 8; ++v) {
            int m = m0 + v + 8 * half;
            size_t idx = (size_t)m * N + n;
            float val = accs[nt][v] + bb;
            if (mode == 1)      val *= g;
            else if (mode == 2) val = (val + extra[idx]) * 0.5f;
            out[idx] = val;
        }
    }
}

// ---------------- sequential scan: h_t = a*h_{t-1} + xh_t  (fp32 carry) ----------
__global__ __launch_bounds__(256) void scan_kernel(const float* __restrict__ xh,
                                                   const float* __restrict__ a_logit,
                                                   const float* __restrict__ h_prev,
                                                   __bf16* __restrict__ hb,
                                                   float* __restrict__ h_last) {
    int idx = blockIdx.x * 256 + threadIdx.x;   // 0 .. BB*DH-1
    if (idx >= BB * DH) return;
    int b = idx >> 11;                          // / DH
    int d = idx & (DH - 1);
    float a = 1.0f / (1.0f + __expf(-a_logit[d]));
    float carry = h_prev[idx];
    const float* xp = xh + (size_t)b * SS * DH + d;
    __bf16*      hp = hb + (size_t)b * SS * DH + d;
    for (int t = 0; t < SS; ++t) {
        carry = a * carry + xp[(size_t)t * DH];
        hp[(size_t)t * DH] = (__bf16)carry;
    }
    h_last[idx] = carry;
}

// ---------------- launcher ----------------
extern "C" void kernel_launch(void* const* d_in, const int* in_sizes, int n_in,
                              void* d_out, int out_size, void* d_ws, size_t ws_size,
                              hipStream_t stream) {
    const float* x       = (const float*)d_in[0];
    const float* h_prev  = (const float*)d_in[1];
    const float* a_logit = (const float*)d_in[2];
    const float* W_dx    = (const float*)d_in[3];
    const float* b_dx    = (const float*)d_in[4];
    const float* W_x     = (const float*)d_in[5];
    const float* b_x     = (const float*)d_in[6];
    const float* W_o     = (const float*)d_in[7];
    const float* b_o     = (const float*)d_in[8];
    float* out = (float*)d_out;                         // (B,S,DOUT) fp32
    float* h_last = out + (size_t)MROWS * DOUT;         // (B,DH) fp32 tail

    // workspace layout (all offsets 256B-aligned)
    char* ws = (char*)d_ws;
    __bf16* xcb  = (__bf16*)(ws);                                    // 8192*1056*2
    __bf16* wdxt = (__bf16*)(ws + 17301504);                         // 1024*1056*2
    __bf16* wxt  = (__bf16*)(ws + 19464192);                         // 2048*1056*2
    __bf16* wot  = (__bf16*)(ws + 23789568);                         // 1024*2048*2
    float*  xh   = (float*) (ws + 27983872);                         // 8192*2048*4
    __bf16* hb   = (__bf16*)(ws + 95092736);                         // 8192*2048*2

    // 1. pack input + positional encoding -> bf16, K-padded
    pack_x_kernel<<<(MROWS * KPAD + 255) / 256, 256, 0, stream>>>(x, xcb);

    // 2. transpose+convert weights -> bf16 W^T with zero K-pad
    pack_w_kernel<<<(DOUT * KPAD + 255) / 256, 256, 0, stream>>>(W_dx, wdxt, DCAT, KPAD, DOUT);
    pack_w_kernel<<<(DH   * KPAD + 255) / 256, 256, 0, stream>>>(W_x,  wxt,  DCAT, KPAD, DH);
    pack_w_kernel<<<(DOUT * DH   + 255) / 256, 256, 0, stream>>>(W_o,  wot,  DH,   DH,   DOUT);

    // 3. dx = xc @ W_dx + b_dx  -> d_out (fp32)
    gemm_bf16_kernel<<<dim3(DOUT / 64, MROWS / 128), 256, 0, stream>>>(
        xcb, wdxt, KPAD, DOUT, out, b_dx, nullptr, 0);

    // 4. xh = (xc @ W_x + b_x) * sqrt(1 - a^2) -> ws (fp32)
    gemm_bf16_kernel<<<dim3(DH / 64, MROWS / 128), 256, 0, stream>>>(
        xcb, wxt, KPAD, DH, xh, b_x, a_logit, 1);

    // 5. recurrent scan: h (bf16) + h_last (fp32)
    scan_kernel<<<(BB * DH + 255) / 256, 256, 0, stream>>>(xh, a_logit, h_prev, hb, h_last);

    // 6. out = (dx + h @ W_o + b_o) / 2  (reads dx from d_out, overwrites in place)
    gemm_bf16_kernel<<<dim3(DOUT / 64, MROWS / 128), 256, 0, stream>>>(
        hb, wot, DH, DOUT, out, b_o, out, 2);
}
